// GraphNet_12163347383045
// MI455X (gfx1250) — matrix operand order repaired and checked by reading, not compiled, not run on hardware
//
#include <hip/hip_runtime.h>
#include <hip/hip_bf16.h>

typedef __attribute__((ext_vector_type(16))) _Float16 v16h;
typedef __attribute__((ext_vector_type(8)))  float    v8f;

// ---------------- problem constants (from reference) ----------------
#define N_CONT   200000
#define N_FEAT   128
#define N_CAT    4
#define MAX_CLS  64
#define N_EDGES  3200000
#define N_GRAPHS 1024
#define H1       64
#define H2       32
#define NCLS     16
#define N_NODES  (N_CONT + N_CAT)       // 200004
#define N_TILES  ((N_NODES + 15) / 16)  // 12501
#define NPAD     200064                 // >= N_TILES*16, multiple of 64

// workspace layout (float offsets; every region 256B aligned)
#define OFF_DINV 0u
#define OFF_EMB  (OFF_DINV + NPAD)                  // 512 floats
#define OFF_A    (OFF_EMB + 512u)                   // xw1 [NPAD x 64], later xw2 [NPAD x 32]
#define OFF_B    (OFF_A + (size_t)NPAD * 64u)       // h1/out1 [NPAD x 64], later out2 [NPAD x 32]
#define OFF_SUM  (OFF_B + (size_t)NPAD * 64u)       // pooled sums [1024 x 32]
#define OFF_CNT  (OFF_SUM + (size_t)N_GRAPHS * H2)  // 1024 floats
#define OFF_W1H  (OFF_CNT + 1024u)                  // W1 as f16: 8192 halves = 4096 floats
#define OFF_W2H  (OFF_W1H + 4096u)                  // W2 as f16: 2048 halves = 1024 floats

// ---------------- weight pre-conversion to f16 (once per launch) ----------
__global__ void k_cvt_w(const float* __restrict__ W1, const float* __restrict__ W2,
                        _Float16* __restrict__ W1h, _Float16* __restrict__ W2h) {
  int idx = blockIdx.x * blockDim.x + threadIdx.x;
  if (idx < N_FEAT * H1) W1h[idx] = (_Float16)W1[idx];
  else if (idx < N_FEAT * H1 + H1 * H2) {
    int j = idx - N_FEAT * H1;
    W2h[j] = (_Float16)W2[j];
  }
}

// ---------------- tiny embedding: emb[c][f] = cat_x[c]·W_emb[c][f] + b_emb ----
__global__ void k_emb(const float* __restrict__ cat_x, const float* __restrict__ W_emb,
                      const float* __restrict__ b_emb, float* __restrict__ emb) {
  int idx = blockIdx.x * blockDim.x + threadIdx.x;
  if (idx >= N_CAT * N_FEAT) return;
  int c = idx / N_FEAT, f = idx % N_FEAT;
  const float* w = W_emb + ((size_t)c * N_FEAT + f) * MAX_CLS;
  const float* x = cat_x + (size_t)c * MAX_CLS;
  float s = b_emb[idx];
  #pragma unroll 8
  for (int m = 0; m < MAX_CLS; ++m) s += w[m] * x[m];
  emb[idx] = s;
}

// ---------------- degree / normalization ----------------
__global__ void k_deg_init(float* __restrict__ deg) {
  int i = blockIdx.x * blockDim.x + threadIdx.x;
  if (i < N_NODES) deg[i] = 1.0f;   // self-loop
}
__global__ void k_deg_count(const int* __restrict__ dst, float* __restrict__ deg) {
  int e = blockIdx.x * blockDim.x + threadIdx.x;
  if (e < N_EDGES) unsafeAtomicAdd(&deg[dst[e]], 1.0f);
}
__global__ void k_dinv(float* __restrict__ deg) {
  int i = blockIdx.x * blockDim.x + threadIdx.x;
  if (i < N_NODES) deg[i] = rsqrtf(deg[i]);  // deg >= 1 always
}

// ---------------- GEMM1: xw1 = X[N x 128] @ W1[128 x 64] via WMMA f16 ----------
__global__ void k_gemm1(const float* __restrict__ num_x, const float* __restrict__ emb,
                        const _Float16* __restrict__ W1h, float* __restrict__ xw1) {
  int lane = threadIdx.x & 31;
  int wave = threadIdx.x >> 5;
  int tile = blockIdx.x * 8 + wave;             // 8 waves / block
  if (tile >= N_TILES) return;                  // wave-uniform exit
  int half = lane >> 4;
  int m0   = tile * 16;
  int mrow = m0 + (lane & 15);
  int r    = mrow < N_NODES ? mrow : N_NODES - 1;
  const float* xrow = (r < N_CONT) ? (num_x + (size_t)r * N_FEAT)
                                   : (emb + (size_t)(r - N_CONT) * N_FEAT);
  v8f acc0 = {}, acc1 = {}, acc2 = {}, acc3 = {};
  for (int k0 = 0; k0 < N_FEAT; k0 += 32) {
    // A tile 16x32 f16: lane<16 -> K in {0..7,16..23}; lane>=16 -> {8..15,24..31}
    v16h a;
    #pragma unroll
    for (int e = 0; e < 16; ++e) {
      int kl = e + (half << 3) + ((e >= 8) ? 8 : 0);
      a[e] = (_Float16)xrow[k0 + kl];
    }
    // B tiles 32x16 f16: lane&15 = N column, element = K (same order as A)
    v16h b0, b1, b2, b3;
    #pragma unroll
    for (int e = 0; e < 16; ++e) {
      int kl = e + (half << 3) + ((e >= 8) ? 8 : 0);
      const _Float16* wrow = W1h + (size_t)(k0 + kl) * H1 + (lane & 15);
      b0[e] = wrow[0];
      b1[e] = wrow[16];
      b2[e] = wrow[32];
      b3[e] = wrow[48];
    }
    acc0 = __builtin_amdgcn_wmma_f32_16x16x32_f16(false, a, false, b0, (short)0, acc0, false, false);
    acc1 = __builtin_amdgcn_wmma_f32_16x16x32_f16(false, a, false, b1, (short)0, acc1, false, false);
    acc2 = __builtin_amdgcn_wmma_f32_16x16x32_f16(false, a, false, b2, (short)0, acc2, false, false);
    acc3 = __builtin_amdgcn_wmma_f32_16x16x32_f16(false, a, false, b3, (short)0, acc3, false, false);
  }
  // D layout: lane n = lane&15; VGPR v -> row m0 + v + 8*half
  int n = lane & 15;
  #pragma unroll
  for (int v = 0; v < 8; ++v) {
    int m = m0 + v + 8 * half;
    if (m < N_NODES) {
      float* o = xw1 + (size_t)m * H1 + n;
      o[0]  = acc0[v];
      o[16] = acc1[v];
      o[32] = acc2[v];
      o[48] = acc3[v];
    }
  }
}

// ---------------- out1 init: bias + self-loop term ----------------
__global__ void k_init_out1(const float* __restrict__ xw1, const float* __restrict__ dinv,
                            const float* __restrict__ b1, float* __restrict__ out1) {
  long long idx = (long long)blockIdx.x * blockDim.x + threadIdx.x;
  if (idx >= (long long)N_NODES * H1) return;
  int i = (int)(idx >> 6), f = (int)(idx & 63);
  float di = dinv[i];
  out1[idx] = b1[f] + xw1[idx] * di * di;
}

// ---------------- edge scatter 1: 16 threads/edge, float4 each ----------------
__global__ void k_scatter1(const int* __restrict__ src, const int* __restrict__ dst,
                           const float* __restrict__ dinv, const float* __restrict__ xw1,
                           float* __restrict__ out1) {
  long long tid = (long long)blockIdx.x * blockDim.x + threadIdx.x;
  long long e = tid >> 4;
  if (e >= N_EDGES) return;
  int q = (int)(tid & 15);
  int s = src[e], d = dst[e];
  float nrm = dinv[s] * dinv[d];
  const float4 v = *((const float4*)(xw1 + (size_t)s * H1) + q);
  float* o = out1 + (size_t)d * H1 + q * 4;
  unsafeAtomicAdd(o + 0, v.x * nrm);
  unsafeAtomicAdd(o + 1, v.y * nrm);
  unsafeAtomicAdd(o + 2, v.z * nrm);
  unsafeAtomicAdd(o + 3, v.w * nrm);
}

// ---------------- GEMM2: xw2 = relu(h1)[N x 64] @ W2[64 x 32] ----------------
__global__ void k_gemm2(const float* __restrict__ h1, const _Float16* __restrict__ W2h,
                        float* __restrict__ xw2) {
  int lane = threadIdx.x & 31;
  int wave = threadIdx.x >> 5;
  int tile = blockIdx.x * 8 + wave;
  if (tile >= N_TILES) return;
  int half = lane >> 4;
  int m0   = tile * 16;
  int mrow = m0 + (lane & 15);
  int r    = mrow < N_NODES ? mrow : N_NODES - 1;
  const float* xrow = h1 + (size_t)r * H1;
  v8f acc0 = {}, acc1 = {};
  for (int k0 = 0; k0 < H1; k0 += 32) {
    v16h a;
    #pragma unroll
    for (int e = 0; e < 16; ++e) {
      int kl = e + (half << 3) + ((e >= 8) ? 8 : 0);
      a[e] = (_Float16)fmaxf(xrow[k0 + kl], 0.0f);   // fused ReLU
    }
    v16h b0, b1;
    #pragma unroll
    for (int e = 0; e < 16; ++e) {
      int kl = e + (half << 3) + ((e >= 8) ? 8 : 0);
      const _Float16* wrow = W2h + (size_t)(k0 + kl) * H2 + (lane & 15);
      b0[e] = wrow[0];
      b1[e] = wrow[16];
    }
    acc0 = __builtin_amdgcn_wmma_f32_16x16x32_f16(false, a, false, b0, (short)0, acc0, false, false);
    acc1 = __builtin_amdgcn_wmma_f32_16x16x32_f16(false, a, false, b1, (short)0, acc1, false, false);
  }
  int n = lane & 15;
  #pragma unroll
  for (int v = 0; v < 8; ++v) {
    int m = m0 + v + 8 * half;
    if (m < N_NODES) {
      float* o = xw2 + (size_t)m * H2 + n;
      o[0]  = acc0[v];
      o[16] = acc1[v];
    }
  }
}

__global__ void k_init_out2(const float* __restrict__ xw2, const float* __restrict__ dinv,
                            const float* __restrict__ b2, float* __restrict__ out2) {
  long long idx = (long long)blockIdx.x * blockDim.x + threadIdx.x;
  if (idx >= (long long)N_NODES * H2) return;
  int i = (int)(idx >> 5), f = (int)(idx & 31);
  float di = dinv[i];
  out2[idx] = b2[f] + xw2[idx] * di * di;
}

// ---------------- edge scatter 2: 8 threads/edge, float4 each ----------------
__global__ void k_scatter2(const int* __restrict__ src, const int* __restrict__ dst,
                           const float* __restrict__ dinv, const float* __restrict__ xw2,
                           float* __restrict__ out2) {
  long long tid = (long long)blockIdx.x * blockDim.x + threadIdx.x;
  long long e = tid >> 3;
  if (e >= N_EDGES) return;
  int q = (int)(tid & 7);
  int s = src[e], d = dst[e];
  float nrm = dinv[s] * dinv[d];
  const float4 v = *((const float4*)(xw2 + (size_t)s * H2) + q);
  float* o = out2 + (size_t)d * H2 + q * 4;
  unsafeAtomicAdd(o + 0, v.x * nrm);
  unsafeAtomicAdd(o + 1, v.y * nrm);
  unsafeAtomicAdd(o + 2, v.z * nrm);
  unsafeAtomicAdd(o + 3, v.w * nrm);
}

// ---------------- pooling ----------------
__global__ void k_pool_zero(float* __restrict__ sums, float* __restrict__ cnt) {
  int i = blockIdx.x * blockDim.x + threadIdx.x;
  if (i < N_GRAPHS * H2) sums[i] = 0.0f;
  if (i < N_GRAPHS) cnt[i] = 0.0f;
}
__global__ void k_count(const int* __restrict__ batch, float* __restrict__ cnt) {
  int i = blockIdx.x * blockDim.x + threadIdx.x;
  if (i < N_NODES) unsafeAtomicAdd(&cnt[batch[i]], 1.0f);
}
__global__ void k_pool(const int* __restrict__ batch, const float* __restrict__ out2,
                       float* __restrict__ sums) {
  long long idx = (long long)blockIdx.x * blockDim.x + threadIdx.x;
  if (idx >= (long long)N_NODES * H2) return;
  int i = (int)(idx >> 5), f = (int)(idx & 31);
  int b = batch[i];
  unsafeAtomicAdd(&sums[(size_t)b * H2 + f], out2[idx]);
}

// ---------------- head: pooled @ Wfc + bfc, softmax, center ----------------
__global__ void k_head(const float* __restrict__ sums, const float* __restrict__ cnt,
                       const float* __restrict__ Wfc, const float* __restrict__ bfc,
                       float* __restrict__ out) {
  int g = blockIdx.x * blockDim.x + threadIdx.x;
  if (g >= N_GRAPHS) return;
  float inv = 1.0f / fmaxf(cnt[g], 1.0f);
  float pooled[H2];
  #pragma unroll
  for (int f = 0; f < H2; ++f) pooled[f] = sums[(size_t)g * H2 + f] * inv;
  float p[NCLS];
  float mx = -1e30f;
  #pragma unroll
  for (int c = 0; c < NCLS; ++c) {
    float l = bfc[c];
    #pragma unroll
    for (int f = 0; f < H2; ++f) l += pooled[f] * Wfc[(size_t)f * NCLS + c];
    p[c] = l;
    mx = fmaxf(mx, l);
  }
  float S = 0.0f;
  #pragma unroll
  for (int c = 0; c < NCLS; ++c) { p[c] = __expf(p[c] - mx); S += p[c]; }
  float invS = 1.0f / S, mean = 0.0f;
  #pragma unroll
  for (int c = 0; c < NCLS; ++c) { p[c] *= invS; mean += p[c]; }
  mean *= (1.0f / NCLS);
  #pragma unroll
  for (int c = 0; c < NCLS; ++c) out[(size_t)g * NCLS + c] = p[c] - mean;
}

// ---------------- launcher ----------------
extern "C" void kernel_launch(void* const* d_in, const int* in_sizes, int n_in,
                              void* d_out, int out_size, void* d_ws, size_t ws_size,
                              hipStream_t stream) {
  const float* num_x = (const float*)d_in[0];
  const float* cat_x = (const float*)d_in[1];
  const float* W_emb = (const float*)d_in[2];
  const float* b_emb = (const float*)d_in[3];
  const float* W1    = (const float*)d_in[4];
  const float* b1    = (const float*)d_in[5];
  const float* W2    = (const float*)d_in[6];
  const float* b2    = (const float*)d_in[7];
  const float* Wfc   = (const float*)d_in[8];
  const float* bfc   = (const float*)d_in[9];
  const int*   eidx  = (const int*)d_in[10];
  const int*   batch = (const int*)d_in[11];
  const int* src = eidx;
  const int* dst = eidx + N_EDGES;

  float* ws   = (float*)d_ws;
  float* dinv = ws + OFF_DINV;
  float* emb  = ws + OFF_EMB;
  float* bufA = ws + OFF_A;   // xw1, later xw2
  float* bufB = ws + OFF_B;   // out1/h1, later out2
  float* sums = ws + OFF_SUM;
  float* cnt  = ws + OFF_CNT;
  _Float16* W1h = (_Float16*)(ws + OFF_W1H);
  _Float16* W2h = (_Float16*)(ws + OFF_W2H);

  // weight f16 conversion, embedding, degree normalization
  k_cvt_w<<<(N_FEAT * H1 + H1 * H2 + 255) / 256, 256, 0, stream>>>(W1, W2, W1h, W2h);
  k_emb<<<(N_CAT * N_FEAT + 63) / 64, 64, 0, stream>>>(cat_x, W_emb, b_emb, emb);
  k_deg_init<<<(N_NODES + 255) / 256, 256, 0, stream>>>(dinv);
  k_deg_count<<<(N_EDGES + 255) / 256, 256, 0, stream>>>(dst, dinv);
  k_dinv<<<(N_NODES + 255) / 256, 256, 0, stream>>>(dinv);

  // conv1: GEMM (WMMA) -> init with bias+self-loop -> edge scatter
  k_gemm1<<<(N_TILES + 7) / 8, 256, 0, stream>>>(num_x, emb, W1h, bufA);
  {
    long long t = (long long)N_NODES * H1;
    k_init_out1<<<(unsigned)((t + 255) / 256), 256, 0, stream>>>(bufA, dinv, b1, bufB);
  }
  {
    long long t = (long long)N_EDGES * 16;
    k_scatter1<<<(unsigned)((t + 255) / 256), 256, 0, stream>>>(src, dst, dinv, bufA, bufB);
  }

  // conv2: GEMM (WMMA, fused ReLU on A) -> init -> edge scatter
  k_gemm2<<<(N_TILES + 7) / 8, 256, 0, stream>>>(bufB, W2h, bufA);     // xw2 overwrites xw1
  {
    long long t = (long long)N_NODES * H2;
    k_init_out2<<<(unsigned)((t + 255) / 256), 256, 0, stream>>>(bufA, dinv, b2, bufB); // out2 overwrites h1
  }
  {
    long long t = (long long)N_EDGES * 8;
    k_scatter2<<<(unsigned)((t + 255) / 256), 256, 0, stream>>>(src, dst, dinv, bufA, bufB);
  }

  // mean-pool + head
  k_pool_zero<<<(N_GRAPHS * H2 + 255) / 256, 256, 0, stream>>>(sums, cnt);
  k_count<<<(N_NODES + 255) / 256, 256, 0, stream>>>(batch, cnt);
  {
    long long t = (long long)N_NODES * H2;
    k_pool<<<(unsigned)((t + 255) / 256), 256, 0, stream>>>(batch, bufB, sums);
  }
  k_head<<<(N_GRAPHS + 127) / 128, 128, 0, stream>>>(sums, cnt, Wfc, bfc, (float*)d_out);
}
